// Net_12532714570351
// MI455X (gfx1250) — matrix operand order repaired and checked by reading, not compiled, not run on hardware
//
#include <hip/hip_runtime.h>
#include <hip/hip_bf16.h>
#include <cstdint>

// ---------------------------------------------------------------------------
// SNN forward (snntorch-style 2-layer LIF), MI455X / gfx1250.
// Phase 1: cur1 = x @ W1^T + b1 via bf16x3-split WMMA, TDM (tensor_load_to_lds)
//          double-buffered LDS staging, TENSORcnt-synchronized.
// Phase 2: 25-step LIF recurrence fully fused, one wave32 per batch row.
// ---------------------------------------------------------------------------

typedef __attribute__((ext_vector_type(16))) __bf16   v16bf;
typedef __attribute__((ext_vector_type(8)))  __bf16   v8bf;
typedef __attribute__((ext_vector_type(8)))  float    v8f;
typedef __attribute__((ext_vector_type(4)))  uint32_t u32x4;
typedef __attribute__((ext_vector_type(8)))  int      i32x8;
typedef __attribute__((ext_vector_type(4)))  int      i32x4;

#define BATCH  4096
#define NI     784
#define KP     800      // K padded to multiple of 32
#define NH     1000
#define NHP    1024     // hidden padded to multiple of 128
#define NO     10
#define NSTEP  25
#define BETA   0.95f
#define THRESH 1.0f

// LDS tile geometry (elements). Row strides padded via TDM pad feature:
// A rows: 32 elems (64B) + 16B pad  -> stride 40 elems (80B, 16B-aligned)
// B rows: 128 elems (256B) + 16B pad -> stride 136 elems (272B, 16B-aligned)
#define LDA    40
#define LDB    136
#define A_ELEMS (32 * LDA)          // 1280
#define B_ELEMS (32 * LDB)          // 4352
#define BUFSZ   (2 * A_ELEMS + 2 * B_ELEMS)   // 11264 elems = 22528 B
#define OFF_AH  0
#define OFF_AL  (A_ELEMS)
#define OFF_BH  (2 * A_ELEMS)
#define OFF_BL  (2 * A_ELEMS + B_ELEMS)

// ---------------- conversion / split kernels -------------------------------

__global__ void k_split_x(const float* __restrict__ x,
                          __bf16* __restrict__ xh, __bf16* __restrict__ xl) {
    int idx = blockIdx.x * blockDim.x + threadIdx.x;
    if (idx >= BATCH * KP) return;
    int row = idx / KP, k = idx - row * KP;
    float v = (k < NI) ? x[row * NI + k] : 0.0f;
    __bf16 h = (__bf16)v;
    xh[idx] = h;
    xl[idx] = (__bf16)(v - (float)h);
}

__global__ void k_split_w1t(const float* __restrict__ W1,
                            __bf16* __restrict__ wh, __bf16* __restrict__ wl) {
    int idx = blockIdx.x * blockDim.x + threadIdx.x;
    if (idx >= KP * NHP) return;
    int k = idx / NHP, n = idx - k * NHP;
    float v = (k < NI && n < NH) ? W1[n * NI + k] : 0.0f;   // transposed: [k][n]
    __bf16 h = (__bf16)v;
    wh[idx] = h;
    wl[idx] = (__bf16)(v - (float)h);
}

__global__ void k_pad_b1(const float* __restrict__ b1, float* __restrict__ b1p) {
    int idx = blockIdx.x * blockDim.x + threadIdx.x;
    if (idx < NHP) b1p[idx] = (idx < NH) ? b1[idx] : 0.0f;
}

// ---------------- TDM descriptor + issue -----------------------------------
// 2D tile load, data_size = 2 bytes. tensor_dims == tile_dims (no OOB ever:
// sources are pre-padded), real row stride in dim0_stride, LDS row padding
// via pad_interval/pad_amount (CDNA5 ISA).
// This toolchain's builtin is the 6-arg clang-23 form:
//   (u32x4 g0, i32x8 g1, i32x4 g2, i32x4 g3, i32x8 extra, i32 cpol)

__device__ __forceinline__ void tdm_load_2d(const __bf16* gsrc, uint32_t lds_byte_off,
                                            uint32_t tile0, uint32_t tile1,
                                            uint32_t stride0_elems,
                                            uint32_t pad_interval, uint32_t pad_amount_enc) {
    uint64_t ga = (uint64_t)(uintptr_t)gsrc;
    u32x4 g0;
    g0[0] = 1u;                                    // count=1, user descriptor
    g0[1] = lds_byte_off;                          // lds_addr
    g0[2] = (uint32_t)ga;                          // global_addr[31:0]
    g0[3] = (uint32_t)((ga >> 32) & 0x01FFFFFFu)   // global_addr[56:32]
          | (2u << 30);                            // type = 2 (image)
    i32x8 g1;
    g1[0] = (int)((1u << 16)                       // data_size = 1 -> 2 bytes
          | (1u << 20)                             // pad_enable
          | (pad_interval << 22)
          | (pad_amount_enc << 25));
    g1[1] = (int)((tile0 & 0xFFFFu) << 16);        // tensor_dim0[15:0] @ bits 63:48
    g1[2] = (int)((tile1 & 0xFFFFu) << 16);        // tensor_dim1[15:0] @ bits 95:80
    g1[3] = (int)((tile0 & 0xFFFFu) << 16);        // tile_dim0 @ bits 127:112
    g1[4] = (int)(tile1 & 0xFFFFu);                // tile_dim1 @ bits 143:128
    g1[5] = (int)stride0_elems;                    // tensor_dim0_stride[31:0]
    g1[6] = 0;                                     // stride0[47:32], stride1 lo
    g1[7] = 0;
    i32x4 gz4 = {0, 0, 0, 0};
    i32x8 gz8 = {0, 0, 0, 0, 0, 0, 0, 0};
    __builtin_amdgcn_tensor_load_to_lds(g0, g1, gz4, gz4, gz8, 0);
}

// ---------------- WMMA GEMM: cur1 = Xs @ W1Ts + b1 -------------------------

__device__ __forceinline__ v8f wmma_bf16(v16bf a, v16bf b, v8f c) {
    return __builtin_amdgcn_wmma_f32_16x16x32_bf16(
        false, a, false, b, (short)0, c, false, false);
}

// A fragment (16x32 bf16) from LDS tile, row stride LDA.
__device__ __forceinline__ v16bf frag_a(const __bf16* tile, int mi, int lane) {
    int row  = mi * 16 + (lane & 15);
    int koff = (lane >> 4) * 8;
    const __bf16* p = tile + row * LDA + koff;
    v8bf lo = *(const v8bf*)(p);
    v8bf hi = *(const v8bf*)(p + 16);
    return __builtin_shufflevector(lo, hi, 0,1,2,3,4,5,6,7,8,9,10,11,12,13,14,15);
}

// B fragment (32x16 bf16) from LDS tile [k][n], row stride LDB; lane l = N col l%16, k = 8*(l/16)+s (s<8), 16+8*(l/16)+(s-8) (s>=8).
__device__ __forceinline__ v16bf frag_b(const __bf16* tile, int nlocal, int lane) {
    const __bf16* p = tile + ((lane >> 4) * 8) * LDB + nlocal + (lane & 15);
    v16bf r;
    #pragma unroll
    for (int s = 0; s < 8; ++s) {
        r[s]     = p[s * LDB];
        r[s + 8] = p[(s + 16) * LDB];
    }
    return r;
}

__global__ __launch_bounds__(128)
void k_gemm1(const __bf16* __restrict__ xh, const __bf16* __restrict__ xl,
             const __bf16* __restrict__ wh, const __bf16* __restrict__ wl,
             const float* __restrict__ b1p, float* __restrict__ cur1) {
    __shared__ __align__(16) __bf16 smem[2 * BUFSZ];   // 45056 B, double-buffered

    const int lane   = threadIdx.x & 31;
    const int wave   = threadIdx.x >> 5;
    const bool loader = (wave == 0);                   // wave-uniform
    const int mbase0 = blockIdx.x * 32;                // 128 M tiles
    const int nblock = blockIdx.y * 128;               // 8 N groups
    const int nloc   = wave * 32;                      // this wave's N within block

    const uint32_t lds_base = (uint32_t)(uintptr_t)(&smem[0]);

    // Prologue: stage kb=0 into buffer 0.
    if (loader) {
        tdm_load_2d(xh + mbase0 * KP,  lds_base + OFF_AH * 2, 32, 32, KP,  3, 3);
        tdm_load_2d(xl + mbase0 * KP,  lds_base + OFF_AL * 2, 32, 32, KP,  3, 3);
        tdm_load_2d(wh + nblock,       lds_base + OFF_BH * 2, 128, 32, NHP, 5, 3);
        tdm_load_2d(wl + nblock,       lds_base + OFF_BL * 2, 128, 32, NHP, 5, 3);
        __builtin_amdgcn_s_wait_tensorcnt(0);
    }
    __syncthreads();

    v8f acc[2][2];
    const v8f vzero = {0.f,0.f,0.f,0.f,0.f,0.f,0.f,0.f};
    acc[0][0] = vzero; acc[0][1] = vzero; acc[1][0] = vzero; acc[1][1] = vzero;

    int cur = 0;
    for (int kb = 0; kb < KP; kb += 32) {
        // Kick off next k-slab into the other buffer (overlaps with compute).
        if (loader && (kb + 32 < KP)) {
            uint32_t nb = lds_base + (uint32_t)((cur ^ 1) * BUFSZ) * 2;
            tdm_load_2d(xh + mbase0 * KP + (kb + 32),      nb + OFF_AH * 2, 32, 32, KP,  3, 3);
            tdm_load_2d(xl + mbase0 * KP + (kb + 32),      nb + OFF_AL * 2, 32, 32, KP,  3, 3);
            tdm_load_2d(wh + (kb + 32) * NHP + nblock,     nb + OFF_BH * 2, 128, 32, NHP, 5, 3);
            tdm_load_2d(wl + (kb + 32) * NHP + nblock,     nb + OFF_BL * 2, 128, 32, NHP, 5, 3);
        }

        const __bf16* buf = &smem[cur * BUFSZ];
        v16bf ah0 = frag_a(buf + OFF_AH, 0, lane);
        v16bf ah1 = frag_a(buf + OFF_AH, 1, lane);
        v16bf al0 = frag_a(buf + OFF_AL, 0, lane);
        v16bf al1 = frag_a(buf + OFF_AL, 1, lane);
        v16bf bh0 = frag_b(buf + OFF_BH, nloc,      lane);
        v16bf bh1 = frag_b(buf + OFF_BH, nloc + 16, lane);
        v16bf bl0 = frag_b(buf + OFF_BL, nloc,      lane);
        v16bf bl1 = frag_b(buf + OFF_BL, nloc + 16, lane);

        // hi*hi, hi*lo, lo*hi, cycling accumulators so consecutive WMMAs
        // never reuse the same D registers.
        acc[0][0] = wmma_bf16(ah0, bh0, acc[0][0]);
        acc[0][1] = wmma_bf16(ah0, bh1, acc[0][1]);
        acc[1][0] = wmma_bf16(ah1, bh0, acc[1][0]);
        acc[1][1] = wmma_bf16(ah1, bh1, acc[1][1]);

        acc[0][0] = wmma_bf16(ah0, bl0, acc[0][0]);
        acc[0][1] = wmma_bf16(ah0, bl1, acc[0][1]);
        acc[1][0] = wmma_bf16(ah1, bl0, acc[1][0]);
        acc[1][1] = wmma_bf16(ah1, bl1, acc[1][1]);

        acc[0][0] = wmma_bf16(al0, bh0, acc[0][0]);
        acc[0][1] = wmma_bf16(al0, bh1, acc[0][1]);
        acc[1][0] = wmma_bf16(al1, bh0, acc[1][0]);
        acc[1][1] = wmma_bf16(al1, bh1, acc[1][1]);

        // Next buffer's DMA must be complete; all waves done reading `cur`
        // before it is overwritten next iteration.
        if (loader) __builtin_amdgcn_s_wait_tensorcnt(0);
        __syncthreads();
        cur ^= 1;
    }

    // C/D layout: VGPR r -> row = r + 8*(lane>>4), col = lane&15
    #pragma unroll
    for (int mi = 0; mi < 2; ++mi) {
        #pragma unroll
        for (int ni = 0; ni < 2; ++ni) {
            int col   = nblock + nloc + ni * 16 + (lane & 15);
            float bb  = b1p[col];
            int rbase = mbase0 + mi * 16 + ((lane >> 4) * 8);
            #pragma unroll
            for (int r = 0; r < 8; ++r)
                cur1[(rbase + r) * NHP + col] = acc[mi][ni][r] + bb;
        }
    }
}

// ---------------- fused 25-step LIF recurrence -----------------------------

__global__ __launch_bounds__(256)
void k_lif(const float* __restrict__ cur1, const float* __restrict__ W2,
           const float* __restrict__ b2,
           float* __restrict__ outS, float* __restrict__ outM) {
    __shared__ float sW2T[NHP * NO];   // 40 KB, [h][o]
    const int tid = threadIdx.x;
    for (int idx = tid; idx < NHP * NO; idx += 256) {
        int h = idx / NO, o = idx - h * NO;
        sW2T[idx] = (h < NH) ? W2[o * NH + h] : 0.0f;
    }
    __syncthreads();

    const int lane = tid & 31;
    const int b    = blockIdx.x * 8 + (tid >> 5);

    float c1[32], m1[32];
    #pragma unroll
    for (int i = 0; i < 32; ++i) {
        c1[i] = cur1[b * NHP + lane + 32 * i];   // padded cols are exactly 0
        m1[i] = 0.0f;
    }
    float m2[NO], b2v[NO];
    #pragma unroll
    for (int o = 0; o < NO; ++o) { m2[o] = 0.0f; b2v[o] = b2[o]; }

    for (int t = 0; t < NSTEP; ++t) {
        float acc[NO];
        #pragma unroll
        for (int o = 0; o < NO; ++o) acc[o] = 0.0f;

        #pragma unroll
        for (int i = 0; i < 32; ++i) {
            float m = m1[i];
            float reset = (m > THRESH) ? THRESH : 0.0f;   // spike(prev mem1 - 1)
            m = BETA * m + c1[i] - reset;
            m1[i] = m;
            if (m > THRESH) {                              // spk1 == 1
                const float* wrow = &sW2T[(lane + 32 * i) * NO];
                #pragma unroll
                for (int o = 0; o < NO; ++o) acc[o] += wrow[o];
            }
        }

        #pragma unroll
        for (int o = 0; o < NO; ++o) {
            float a = acc[o];
            a += __shfl_xor(a, 16, 32);
            a += __shfl_xor(a,  8, 32);
            a += __shfl_xor(a,  4, 32);
            a += __shfl_xor(a,  2, 32);
            a += __shfl_xor(a,  1, 32);
            float mm = m2[o];
            float r2 = (mm > THRESH) ? THRESH : 0.0f;      // spike(prev mem2 - 1)
            mm = BETA * mm + (a + b2v[o]) - r2;
            m2[o] = mm;
        }

        if (lane == 0) {
            long base = ((long)t * BATCH + b) * NO;
            #pragma unroll
            for (int o = 0; o < NO; ++o) {
                outS[base + o] = (m2[o] > THRESH) ? 1.0f : 0.0f;
                outM[base + o] = m2[o];
            }
        }
    }
}

// ---------------- launch ----------------------------------------------------

extern "C" void kernel_launch(void* const* d_in, const int* in_sizes, int n_in,
                              void* d_out, int out_size, void* d_ws, size_t ws_size,
                              hipStream_t stream) {
    const float* x  = (const float*)d_in[0];
    const float* W1 = (const float*)d_in[1];
    const float* b1 = (const float*)d_in[2];
    const float* W2 = (const float*)d_in[3];
    const float* b2 = (const float*)d_in[4];
    (void)in_sizes; (void)n_in; (void)out_size; (void)ws_size;

    char* ws = (char*)d_ws;
    size_t off = 0;
    auto alloc = [&](size_t bytes) {
        void* p = ws + off;
        off = (off + bytes + 255) & ~(size_t)255;
        return p;
    };
    __bf16* xh  = (__bf16*)alloc((size_t)BATCH * KP * 2);
    __bf16* xl  = (__bf16*)alloc((size_t)BATCH * KP * 2);
    __bf16* wh  = (__bf16*)alloc((size_t)KP * NHP * 2);
    __bf16* wl  = (__bf16*)alloc((size_t)KP * NHP * 2);
    float*  b1p = (float*)alloc((size_t)NHP * 4);
    float*  cur1= (float*)alloc((size_t)BATCH * NHP * 4);   // ~33 MB total ws

    k_split_x  <<<(BATCH * KP + 255) / 256, 256, 0, stream>>>(x, xh, xl);
    k_split_w1t<<<(KP * NHP  + 255) / 256, 256, 0, stream>>>(W1, wh, wl);
    k_pad_b1   <<<(NHP + 255) / 256, 256, 0, stream>>>(b1, b1p);

    k_gemm1<<<dim3(BATCH / 32, NHP / 128), 128, 0, stream>>>(xh, xl, wh, wl, b1p, cur1);

    float* outS = (float*)d_out;
    float* outM = outS + (size_t)NSTEP * BATCH * NO;
    k_lif<<<BATCH / 8, 256, 0, stream>>>(cur1, W2, b2, outS, outM);
}
